// OCRSlotAttention_29772713296513
// MI455X (gfx1250) — compile-verified, matrix operand order checked by reading, not run on hardware
//
#include <hip/hip_runtime.h>
#include <hip/hip_bf16.h>

typedef __attribute__((ext_vector_type(16))) _Float16 v16h;
typedef __attribute__((ext_vector_type(8)))  float    v8f;
typedef _Float16 half_t;

#define B_     64
#define NKV    1024
#define NQ     8
#define S_     256
#define HM_    512
#define EPS_   1e-8f
#define LNE_   1e-5f

// ---------------- WMMA fragment helpers (CDNA5 16x16x32 f16 layout) ----------
// 16-bit operand layout: lane L, klo = (L>=16)?8:0; elems 0..7 hold K =
// klo..klo+7, elems 8..15 hold K = klo+16..klo+23 (row/col index = L&15).

// A-fragment from row-major source (ld in halves): two 16B vector loads.
__device__ inline v16h frag_rowmajor(const half_t* p, int ld, int lane) {
  int m   = lane & 15;
  int klo = (lane >> 4) << 3;
  const half_t* q = p + m * ld + klo;
  union { uint4 u[2]; v16h h; } c;
  c.u[0] = *(const uint4*)(q);
  c.u[1] = *(const uint4*)(q + 16);
  return c.h;
}
// Fragment from pre-packed storage: packed[...][lane][16] -> one 32B load.
__device__ inline v16h frag_packed(const half_t* p, int lane) {
  const half_t* q = p + lane * 16;
  union { uint4 u[2]; v16h h; } c;
  c.u[0] = *(const uint4*)(q);
  c.u[1] = *(const uint4*)(q + 8);
  return c.h;
}
// (lane, elem) -> K-within-32
__device__ inline int klocal_of(int lane, int e) {
  int klo = (lane >> 4) << 3;
  return (e < 8) ? (klo + e) : (klo + 8 + e);
}
// inverse: K-within-32 -> (lane group offset 0/16, elem)
__device__ inline void inv_klocal(int k_local, int& laneoff, int& e) {
  laneoff = ((k_local & 15) >= 8) ? 16 : 0;
  e = (k_local & 7) + ((k_local >= 16) ? 8 : 0);
}
__device__ inline v8f wmma_f16(v16h a, v16h b, v8f c) {
  return __builtin_amdgcn_wmma_f32_16x16x32_f16(false, a, false, b, (short)0, c,
                                                false, false);
}

// ---------------- Wk/Wv fp32 -> fragment-packed fp16 -------------------------
// packed idx = ((kk*16 + nt)*32 + lane)*16 + e ; 8 kk x 16 nt tiles
__global__ void k_cvt(const float* __restrict__ wk, const float* __restrict__ wv,
                      half_t* __restrict__ wkp, half_t* __restrict__ wvp) {
  int idx  = blockIdx.x * 256 + threadIdx.x;  // 65536 halves per matrix
  int e    = idx & 15;
  int lane = (idx >> 4) & 31;
  int nt   = (idx >> 9) & 15;
  int kk   = idx >> 13;
  int k = kk * 32 + klocal_of(lane, e);
  int n = nt * 16 + (lane & 15);
  wkp[idx] = (half_t)wk[k * 256 + n];
  wvp[idx] = (half_t)wv[k * 256 + n];
}

// ---------------- fused LayerNorm(inputs) + K/V projection (WMMA) ------------
// grid: (B*NKV)/32 blocks, 256 threads (8 waves: 2 M-tiles x 4 N-quadrants)
// K written dense row-major (A-operand of logits); V written fragment-packed
// (B-operand of attn@V): vpk[b][krow=32][nt=16][lane=32][e=16]
__global__ void __launch_bounds__(256) k_proj(
    const float* __restrict__ x, const float* __restrict__ g,
    const float* __restrict__ bb, const half_t* __restrict__ wkp,
    const half_t* __restrict__ wvp, half_t* __restrict__ kf,
    half_t* __restrict__ vpk) {
  __shared__ float xs[32][256];
  __shared__ __align__(32) half_t xh[32][256];
  __shared__ float psum[32][8], psq[32][8];
  __shared__ float mu[32], rs[32];
  int t = threadIdx.x;
  long base = (long)blockIdx.x * 32 * 256;
  for (int i = t; i < 32 * 256; i += 256) xs[i >> 8][i & 255] = x[base + i];
  __syncthreads();
  {
    int row = t >> 3, seg = t & 7;
    float s = 0.f, s2 = 0.f;
    for (int c = seg * 32; c < seg * 32 + 32; ++c) {
      float v_ = xs[row][c]; s += v_; s2 += v_ * v_;
    }
    psum[row][seg] = s; psq[row][seg] = s2;
  }
  __syncthreads();
  if (t < 32) {
    float s = 0.f, s2 = 0.f;
    for (int j = 0; j < 8; ++j) { s += psum[t][j]; s2 += psq[t][j]; }
    float m = s * (1.f / 256.f);
    mu[t] = m; rs[t] = rsqrtf(s2 * (1.f / 256.f) - m * m + LNE_);
  }
  __syncthreads();
  for (int i = t; i < 32 * 256; i += 256) {
    int row = i >> 8, col = i & 255;
    xh[row][col] = (half_t)((xs[row][col] - mu[row]) * rs[row] * g[col] + bb[col]);
  }
  __syncthreads();

  int lane = t & 31, w = t >> 5;
  int mt = w & 1, nq4 = w >> 1;
  v8f ak[4], av[4];
  for (int i = 0; i < 4; ++i)
    for (int j = 0; j < 8; ++j) { ak[i][j] = 0.f; av[i][j] = 0.f; }
  for (int kk = 0; kk < 8; ++kk) {
    // issue ALL fragment loads of this K-step first (independent b128s),
    // then the WMMA burst -> backend can use partial s_wait_loadcnt
    v16h a = frag_rowmajor(&xh[mt * 16][kk * 32], 256, lane);
    v16h bk[4], bv[4];
#pragma unroll
    for (int ti = 0; ti < 4; ++ti) {
      int nt = nq4 * 4 + ti;
      bk[ti] = frag_packed(wkp + (long)(kk * 16 + nt) * 512, lane);
      bv[ti] = frag_packed(wvp + (long)(kk * 16 + nt) * 512, lane);
    }
#pragma unroll
    for (int ti = 0; ti < 4; ++ti) {
      ak[ti] = wmma_f16(a, bk[ti], ak[ti]);
      av[ti] = wmma_f16(a, bv[ti], av[ti]);
    }
  }
  int b    = blockIdx.x >> 5;   // 32 blocks per batch (1024 rows)
  int krow = blockIdx.x & 31;   // which 32-row band (K index for attn@V)
  half_t* vb = vpk + (long)b * 262144;
  long rb = (long)blockIdx.x * 32 + mt * 16;
  int mo = (lane >> 4) << 3;
  int nl = lane & 15;
#pragma unroll
  for (int ti = 0; ti < 4; ++ti) {
    int nt  = nq4 * 4 + ti;
    int col = nt * 16 + nl;
#pragma unroll
    for (int r = 0; r < 8; ++r) {
      long row = rb + r + mo;
      kf[row * 256 + col] = (half_t)(ak[ti][r] * 0.125f);  // hd^-0.5
      int k_local = mt * 16 + r + mo;
      int lo, e; inv_klocal(k_local, lo, e);
      vb[((long)(krow * 16 + nt) * 32 + lo + nl) * 16 + e] = (half_t)(av[ti][r]);
    }
  }
}

// ---------------- column sums of V from packed layout ------------------------
__global__ void k_colv(const half_t* __restrict__ vpk, float* __restrict__ colv) {
  int b = blockIdx.x, d = threadIdx.x;
  const half_t* vb = vpk + (long)b * 262144;
  int nt = d >> 4, nl = d & 15;
  float s = 0.f;
  for (int n = 0; n < NKV; ++n) {
    int kk = n >> 5, kl = n & 31;
    int lo, e; inv_klocal(kl, lo, e);
    s += (float)vb[((long)(kk * 16 + nt) * 32 + lo + nl) * 16 + e];
  }
  colv[b * 256 + d] = s;
}

// ---------------- LN(slots) + q = s@Wq, packed block-diagonal Qb -------------
// qbp[b][kk=8][ti=2][lane=32][e=16]
__global__ void __launch_bounds__(256) k_qbuild(
    const float* __restrict__ slots, const float* __restrict__ g,
    const float* __restrict__ bb, const float* __restrict__ wq,
    half_t* __restrict__ qbp) {
  __shared__ float sln[8][256];
  __shared__ float ps[8][32], pq[8][32];
  __shared__ float mu[8], rs[8];
  __shared__ half_t qlds[256][32];
  int b = blockIdx.x, t = threadIdx.x;
  const float* sp = slots + (long)b * NQ * S_;
  {
    int row = t >> 5, seg = t & 31;
    float s = 0.f, s2 = 0.f;
    for (int c = seg * 8; c < seg * 8 + 8; ++c) {
      float v_ = sp[row * 256 + c];
      sln[row][c] = v_; s += v_; s2 += v_ * v_;
    }
    ps[row][seg] = s; pq[row][seg] = s2;
  }
  __syncthreads();
  if (t < 8) {
    float s = 0.f, s2 = 0.f;
    for (int j = 0; j < 32; ++j) { s += ps[t][j]; s2 += pq[t][j]; }
    float m = s * (1.f / 256.f);
    mu[t] = m; rs[t] = rsqrtf(s2 * (1.f / 256.f) - m * m + LNE_);
  }
  __syncthreads();
  for (int i = t; i < 8 * 256; i += 256) {
    int r = i >> 8, c = i & 255;
    sln[r][c] = (sln[r][c] - mu[r]) * rs[r] * g[c] + bb[c];
  }
  __syncthreads();
  float q8[8];
  for (int i = 0; i < 8; ++i) q8[i] = 0.f;
  for (int k = 0; k < 256; ++k) {
    float wv_ = wq[k * 256 + t];
#pragma unroll
    for (int i = 0; i < 8; ++i) q8[i] += sln[i][k] * wv_;
  }
  for (int i = t; i < 256 * 32; i += 256) ((half_t*)qlds)[i] = (half_t)0.f;
  __syncthreads();
  {
    int h = t >> 6;  // head of k-index t
#pragma unroll
    for (int qi = 0; qi < 8; ++qi) qlds[t][h * 8 + qi] = (half_t)q8[qi];
  }
  __syncthreads();
  half_t* qp = qbp + (long)b * 8192;
#pragma unroll
  for (int rep = 0; rep < 2; ++rep) {   // 512 (kk,ti,lane) triples / 256 thr
    int trip = rep * 256 + t;
    int lane = trip & 31, ti = (trip >> 5) & 1, kk = trip >> 6;
    union { uint4 u[2]; half_t h[16]; } c;
#pragma unroll
    for (int e = 0; e < 16; ++e)
      c.h[e] = qlds[kk * 32 + klocal_of(lane, e)][ti * 16 + (lane & 15)];
    *(uint4*)(qp + trip * 16)     = c.u[0];
    *(uint4*)(qp + trip * 16 + 8) = c.u[1];
  }
}

// ---------------- logits GEMM (WMMA) + fused joint softmax -------------------
// grid: B * (NKV/128); attn written fragment-packed for attn@V A-operand:
// apk[b][kk=32][mt=2][lane=32][e=16]
__global__ void __launch_bounds__(256) k_logits(
    const half_t* __restrict__ kf, const half_t* __restrict__ qbp,
    half_t* __restrict__ apk, float* __restrict__ colsum,
    float* __restrict__ vis_out) {
  __shared__ float lg[128][32];
  __shared__ float csl[32];
  int t = threadIdx.x, lane = t & 31, w = t >> 5;
  int b  = blockIdx.x >> 3;
  int rb = (blockIdx.x & 7) * 128;
  if (t < 32) csl[t] = 0.f;
  const half_t* kp = kf + ((long)b * NKV + rb + w * 16) * 256;
  const half_t* qp = qbp + (long)b * 8192;
  v8f acc[2];
  for (int i = 0; i < 2; ++i)
    for (int j = 0; j < 8; ++j) acc[i][j] = 0.f;
  for (int kk = 0; kk < 8; ++kk) {
    v16h a   = frag_rowmajor(kp + kk * 32, 256, lane);
    v16h bq0 = frag_packed(qp + (long)(kk * 2 + 0) * 512, lane);
    v16h bq1 = frag_packed(qp + (long)(kk * 2 + 1) * 512, lane);
    acc[0] = wmma_f16(a, bq0, acc[0]);
    acc[1] = wmma_f16(a, bq1, acc[1]);
  }
  int mo = (lane >> 4) << 3;
#pragma unroll
  for (int ti = 0; ti < 2; ++ti)
#pragma unroll
    for (int r = 0; r < 8; ++r)
      lg[w * 16 + r + mo][ti * 16 + (lane & 15)] = acc[ti][r];
  __syncthreads();
  if (t < 128) {
    float row[32];
    float mx = -3.0e38f;
#pragma unroll
    for (int c = 0; c < 32; ++c) { row[c] = lg[t][c]; mx = fmaxf(mx, row[c]); }
    float s = 0.f;
#pragma unroll
    for (int c = 0; c < 32; ++c) { row[c] = __expf(row[c] - mx); s += row[c]; }
    float inv = 1.f / s;
    int n = rb + t;
#pragma unroll
    for (int c = 0; c < 32; ++c) { row[c] *= inv; atomicAdd(&csl[c], row[c]); }
    // packed store of attn^T fragments (this row = one K value)
    half_t* ab = apk + (long)b * 32768;
    int kk = n >> 5, kl = n & 31;
    int lo, e; inv_klocal(kl, lo, e);
#pragma unroll
    for (int mt = 0; mt < 2; ++mt)
#pragma unroll
      for (int m = 0; m < 16; ++m)
        ab[((long)(kk * 2 + mt) * 32 + m + lo) * 16 + e] =
            (half_t)row[mt * 16 + m];
#pragma unroll
    for (int qi = 0; qi < 8; ++qi) {  // attn_vis = sum over heads
      float v_ = row[qi] + row[8 + qi] + row[16 + qi] + row[24 + qi];
      vis_out[(long)b * NQ * NKV + qi * NKV + n] = v_;
    }
  }
  __syncthreads();
  if (t < 32) atomicAdd(&colsum[b * 32 + t], csl[t]);
}

__global__ void k_inv(const float* __restrict__ colsum, float* __restrict__ inv) {
  int i = blockIdx.x * 256 + threadIdx.x;
  if (i < B_ * 32) inv[i] = 1.f / (colsum[i] + (float)NKV * EPS_);
}

// ---------------- updates = norm(attn+eps)^T @ V (WMMA, all packed) ----------
__global__ void __launch_bounds__(256) k_attnv(
    const half_t* __restrict__ apk, const half_t* __restrict__ vpk,
    const float* __restrict__ colv, const float* __restrict__ invs,
    float* __restrict__ upd) {
  int t = threadIdx.x, lane = t & 31, w = t >> 5;
  int b = blockIdx.x;
  const half_t* ab = apk + (long)b * 32768;
  const half_t* vb = vpk + (long)b * 262144;
  v8f acc[2][2];
  for (int i = 0; i < 2; ++i)
    for (int j = 0; j < 2; ++j)
      for (int e = 0; e < 8; ++e) acc[i][j][e] = 0.f;
  for (int kk = 0; kk < 32; ++kk) {
    // all four fragment loads first, then the WMMA burst
    v16h a0 = frag_packed(ab + (long)(kk * 2 + 0) * 512, lane);
    v16h a1 = frag_packed(ab + (long)(kk * 2 + 1) * 512, lane);
    v16h b0 = frag_packed(vb + (long)(kk * 16 + 2 * w + 0) * 512, lane);
    v16h b1 = frag_packed(vb + (long)(kk * 16 + 2 * w + 1) * 512, lane);
    acc[0][0] = wmma_f16(a0, b0, acc[0][0]);
    acc[1][0] = wmma_f16(a1, b0, acc[1][0]);
    acc[0][1] = wmma_f16(a0, b1, acc[0][1]);
    acc[1][1] = wmma_f16(a1, b1, acc[1][1]);
  }
  int mo = (lane >> 4) << 3;
#pragma unroll
  for (int mt = 0; mt < 2; ++mt)
#pragma unroll
    for (int ntl = 0; ntl < 2; ++ntl) {
      int d = (2 * w + ntl) * 16 + (lane & 15);
#pragma unroll
      for (int r = 0; r < 8; ++r) {
        int hq = mt * 16 + r + mo;
        int h = hq >> 3;
        if ((d >> 6) == h) {  // keep only this head's 64-wide slice
          int qi = hq & 7;
          float val = (acc[mt][ntl][r] + EPS_ * colv[b * 256 + d]) *
                      invs[b * 32 + hq];
          upd[((long)b * NQ + qi) * S_ + d] = val;
        }
      }
    }
}

// ---------------- GRU cell (fp32 VALU), one batch per block ------------------
__global__ void __launch_bounds__(256) k_gru(
    const float* __restrict__ upd, const float* __restrict__ hprev,
    const float* __restrict__ wih, const float* __restrict__ whh,
    const float* __restrict__ bih, const float* __restrict__ bhh,
    float* __restrict__ hout) {
  __shared__ float u[8][256], hp[8][256];
  int b = blockIdx.x, t = threadIdx.x;
  for (int i = t; i < 8 * 256; i += 256) {
    u[i >> 8][i & 255] = upd[(long)b * 2048 + i];
    hp[i >> 8][i & 255] = hprev[(long)b * 2048 + i];
  }
  __syncthreads();
  float r8[8], z8[8];
  {  // r
    float a0[8], a1[8];
    for (int i = 0; i < 8; ++i) { a0[i] = 0.f; a1[i] = 0.f; }
    const float* wi = wih + (long)t * 256;
    const float* wh = whh + (long)t * 256;
    for (int k = 0; k < 256; ++k) {
      float wi_ = wi[k], wh_ = wh[k];
#pragma unroll
      for (int i = 0; i < 8; ++i) { a0[i] += u[i][k] * wi_; a1[i] += hp[i][k] * wh_; }
    }
    float bsum = bih[t] + bhh[t];
    for (int i = 0; i < 8; ++i) r8[i] = 1.f / (1.f + __expf(-(a0[i] + a1[i] + bsum)));
  }
  {  // z
    float a0[8], a1[8];
    for (int i = 0; i < 8; ++i) { a0[i] = 0.f; a1[i] = 0.f; }
    const float* wi = wih + (long)(256 + t) * 256;
    const float* wh = whh + (long)(256 + t) * 256;
    for (int k = 0; k < 256; ++k) {
      float wi_ = wi[k], wh_ = wh[k];
#pragma unroll
      for (int i = 0; i < 8; ++i) { a0[i] += u[i][k] * wi_; a1[i] += hp[i][k] * wh_; }
    }
    float bsum = bih[256 + t] + bhh[256 + t];
    for (int i = 0; i < 8; ++i) z8[i] = 1.f / (1.f + __expf(-(a0[i] + a1[i] + bsum)));
  }
  {  // n + output
    float a0[8], a1[8];
    for (int i = 0; i < 8; ++i) { a0[i] = 0.f; a1[i] = 0.f; }
    const float* wi = wih + (long)(512 + t) * 256;
    const float* wh = whh + (long)(512 + t) * 256;
    for (int k = 0; k < 256; ++k) {
      float wi_ = wi[k], wh_ = wh[k];
#pragma unroll
      for (int i = 0; i < 8; ++i) { a0[i] += u[i][k] * wi_; a1[i] += hp[i][k] * wh_; }
    }
    float bi3 = bih[512 + t], bh3 = bhh[512 + t];
    for (int i = 0; i < 8; ++i) {
      float n = tanhf(a0[i] + bi3 + r8[i] * (a1[i] + bh3));
      float h = (1.f - z8[i]) * n + z8[i] * hp[i][t];
      hout[(long)b * 2048 + i * 256 + t] = h;
    }
  }
}

// ---------------- LN + MLP residual (fp32 VALU), one batch per block ---------
__global__ void __launch_bounds__(256) k_mlp(
    const float* __restrict__ hin, const float* __restrict__ g,
    const float* __restrict__ bb, const float* __restrict__ w1,
    const float* __restrict__ b1, const float* __restrict__ w2,
    const float* __restrict__ b2, float* __restrict__ slots_out,
    float* __restrict__ dout, int write_out) {
  __shared__ float h[8][256], m[8][256], hid[8][512];
  __shared__ float red[256];
  int b = blockIdx.x, t = threadIdx.x;
  for (int i = t; i < 2048; i += 256) h[i >> 8][i & 255] = hin[(long)b * 2048 + i];
  __syncthreads();
  for (int row = 0; row < 8; ++row) {
    float x = h[row][t];
    red[t] = x; __syncthreads();
    for (int s = 128; s > 0; s >>= 1) { if (t < s) red[t] += red[t + s]; __syncthreads(); }
    float mu = red[0] * (1.f / 256.f); __syncthreads();
    float d = x - mu;
    red[t] = d * d; __syncthreads();
    for (int s = 128; s > 0; s >>= 1) { if (t < s) red[t] += red[t + s]; __syncthreads(); }
    float rs = rsqrtf(red[0] * (1.f / 256.f) + LNE_); __syncthreads();
    m[row][t] = d * rs * g[t] + bb[t];
  }
  __syncthreads();
  for (int half = 0; half < 2; ++half) {
    int col = half * 256 + t;
    float a[8];
    for (int i = 0; i < 8; ++i) a[i] = 0.f;
    for (int k = 0; k < 256; ++k) {
      float w_ = w1[(long)k * 512 + col];
#pragma unroll
      for (int i = 0; i < 8; ++i) a[i] += m[i][k] * w_;
    }
    float bias = b1[col];
    for (int i = 0; i < 8; ++i) hid[i][col] = fmaxf(a[i] + bias, 0.f);
  }
  __syncthreads();
  {
    float a[8];
    for (int i = 0; i < 8; ++i) a[i] = 0.f;
    for (int j = 0; j < 512; ++j) {
      float w_ = w2[(long)j * 256 + t];
#pragma unroll
      for (int i = 0; i < 8; ++i) a[i] += hid[i][j] * w_;
    }
    float bias = b2[t];
    for (int i = 0; i < 8; ++i) {
      float o = h[i][t] + a[i] + bias;
      slots_out[(long)b * 2048 + i * 256 + t] = o;
      if (write_out) dout[(long)b * 2048 + i * 256 + t] = o;
    }
  }
}

// ---------------- host orchestration ----------------------------------------
extern "C" void kernel_launch(void* const* d_in, const int* in_sizes, int n_in,
                              void* d_out, int out_size, void* d_ws,
                              size_t ws_size, hipStream_t stream) {
  (void)in_sizes; (void)n_in; (void)out_size; (void)ws_size;
  const float* inputs   = (const float*)d_in[0];
  const float* slots0   = (const float*)d_in[1];
  const float* ln_in_g  = (const float*)d_in[2];
  const float* ln_in_b  = (const float*)d_in[3];
  const float* ln_sl_g  = (const float*)d_in[4];
  const float* ln_sl_b  = (const float*)d_in[5];
  const float* ln_mlp_g = (const float*)d_in[6];
  const float* ln_mlp_b = (const float*)d_in[7];
  const float* Wq       = (const float*)d_in[8];
  const float* Wk       = (const float*)d_in[9];
  const float* Wv       = (const float*)d_in[10];
  const float* gwih     = (const float*)d_in[11];
  const float* gwhh     = (const float*)d_in[12];
  const float* gbih     = (const float*)d_in[13];
  const float* gbhh     = (const float*)d_in[14];
  const float* w1       = (const float*)d_in[15];
  const float* b1       = (const float*)d_in[16];
  const float* w2       = (const float*)d_in[17];
  const float* b2       = (const float*)d_in[18];

  char* ws = (char*)d_ws;
  half_t* kf   = (half_t*)(ws + 0);          // 32 MB  K dense f16 (pre-scaled)
  half_t* vpk  = (half_t*)(ws + 33554432);   // 32 MB  V fragment-packed
  half_t* wkp  = (half_t*)(ws + 67108864);   // 128 KB Wk packed
  half_t* wvp  = (half_t*)(ws + 67239936);   // 128 KB Wv packed
  float*  colv = (float*)(ws + 67371008);    // 64 KB  column sums of V
  half_t* qbp  = (half_t*)(ws + 67436544);   // 1 MB   block-diag Q packed
  half_t* apk  = (half_t*)(ws + 68485120);   // 4 MB   attn^T fragment-packed
  float*  csum = (float*)(ws + 72679424);    // 8 KB
  float*  invs = (float*)(ws + 72687616);    // 8 KB
  float*  upd  = (float*)(ws + 72695808);    // 512 KB
  float*  scur = (float*)(ws + 73220096);    // 512 KB current slots
  float*  smid = (float*)(ws + 73744384);    // 512 KB post-GRU slots

  float* out_slots = (float*)d_out;
  float* out_vis   = out_slots + (long)B_ * NQ * S_;

  hipMemcpyAsync(scur, slots0, (size_t)B_ * NQ * S_ * sizeof(float),
                 hipMemcpyDeviceToDevice, stream);
  k_cvt<<<256, 256, 0, stream>>>(Wk, Wv, wkp, wvp);
  k_proj<<<(B_ * NKV) / 32, 256, 0, stream>>>(inputs, ln_in_g, ln_in_b, wkp, wvp,
                                              kf, vpk);
  k_colv<<<B_, 256, 0, stream>>>(vpk, colv);

  for (int it = 0; it < 3; ++it) {
    hipMemsetAsync(csum, 0, B_ * 32 * sizeof(float), stream);
    k_qbuild<<<B_, 256, 0, stream>>>(scur, ln_sl_g, ln_sl_b, Wq, qbp);
    k_logits<<<B_ * (NKV / 128), 256, 0, stream>>>(kf, qbp, apk, csum, out_vis);
    k_inv<<<8, 256, 0, stream>>>(csum, invs);
    k_attnv<<<B_, 256, 0, stream>>>(apk, vpk, colv, invs, upd);
    k_gru<<<B_, 256, 0, stream>>>(upd, scur, gwih, gwhh, gbih, gbhh, smid);
    k_mlp<<<B_, 256, 0, stream>>>(smid, ln_mlp_g, ln_mlp_b, w1, b1, w2, b2, scur,
                                  out_slots, (it == 2) ? 1 : 0);
  }
}
